// AttentionBlock_84198538871201
// MI455X (gfx1250) — compile-verified, matrix operand order checked by reading, not run on hardware
//
#include <hip/hip_runtime.h>
#include <hip/hip_bf16.h>

// ---------------------------------------------------------------------------
// AttentionBlock for MI455X (gfx1250): bf16 WMMA GEMMs, wave32,
// async-to-LDS staging of the B operand (double buffered), software-pipelined
// A-fragment loads with a 2x-unrolled ping-pong loop (no rotation copies).
// Shapes fixed: B=32, H=W=32, C=256, HW=1024, N=32768.
// ---------------------------------------------------------------------------

#define BATCH   32
#define CDIM    256
#define HWDIM   1024            // H*W
#define NPIX    32768           // BATCH*HWDIM
#define EPSV    1e-6f
#define SCALE_S 0.0625f         // 256^-0.5

typedef __bf16  bf16_t;
typedef bf16_t  v16bf __attribute__((ext_vector_type(16)));
typedef float   v8f   __attribute__((ext_vector_type(8)));

// float -> bf16 bits, round-to-nearest-even
__device__ __forceinline__ unsigned short f2bf(float x) {
    unsigned int u = __float_as_uint(x);
    unsigned int r = u + 0x7FFFu + ((u >> 16) & 1u);
    return (unsigned short)(r >> 16);
}

__device__ __forceinline__ v8f wmma_bf16(v16bf a, v16bf b, v8f c) {
    // D = A(16x32 bf16) x B(32x16 bf16) + C(16x16 f32)
    return __builtin_amdgcn_wmma_f32_16x16x32_bf16(
        /*neg_a=*/false, a, /*neg_b=*/false, b,
        /*c_mod=*/(short)0, c, /*reuse_a=*/false, /*reuse_b=*/false);
}

// 16x32 bf16 fragment from row-major [rows, ld] bf16(bits) matrix in global.
// Lanes 0-15 hold K chunks {0..7,16..23}, lanes 16-31 {8..15,24..31}.
__device__ __forceinline__ v16bf load_frag_g(const unsigned short* base, int ld,
                                             int row, int k0, int lh) {
    const unsigned short* p = base + (size_t)row * ld + k0 + lh * 8;
    union { uint4 u[2]; v16bf f; } t;
    t.u[0] = *(const uint4*)(p);
    t.u[1] = *(const uint4*)(p + 16);
    return t.f;
}

// Same fragment sourced from an LDS tile [128 rows, 32 cols] bf16.
__device__ __forceinline__ v16bf load_frag_lds(const unsigned short* sh,
                                               int row, int lh) {
    const unsigned short* p = sh + row * 32 + lh * 8;
    union { uint4 u[2]; v16bf f; } t;
    t.u[0] = *(const uint4*)(p);
    t.u[1] = *(const uint4*)(p + 16);
    return t.f;
}

// ---------------------------------------------------------------------------
// Kernel: fold (Wq+Wk+Wv)/3 and transpose weights to [N,K] bf16.
// ---------------------------------------------------------------------------
__global__ void prep_weights_kernel(const float* __restrict__ w_qkv,    // [256,768]
                                    const float* __restrict__ w_out,    // [256,256]
                                    unsigned short* __restrict__ wsumT, // [256,256] (n,k)
                                    unsigned short* __restrict__ woutT) // [256,256] (n,k)
{
    int t = blockIdx.x * blockDim.x + threadIdx.x;   // 65536 threads
    int k = t & 255;
    int n = t >> 8;
    float s = (w_qkv[(size_t)k * 768 + n] +
               w_qkv[(size_t)k * 768 + n + 256] +
               w_qkv[(size_t)k * 768 + n + 512]) * (1.0f / 3.0f);
    wsumT[(size_t)n * 256 + k] = f2bf(s);
    woutT[(size_t)n * 256 + k] = f2bf(w_out[(size_t)k * 256 + n]);
}

// ---------------------------------------------------------------------------
// Kernel: per-pixel GroupNorm over 32 groups of 8 contiguous channels -> bf16.
// One thread per (pixel, group).
// ---------------------------------------------------------------------------
__global__ void groupnorm_kernel(const float* __restrict__ x,
                                 const float* __restrict__ gamma,
                                 const float* __restrict__ beta,
                                 unsigned short* __restrict__ xn)     // [NPIX,256]
{
    int t = blockIdx.x * blockDim.x + threadIdx.x;   // NPIX*32 threads
    int grp = t & 31;
    size_t pix = (size_t)(t >> 5);
    size_t base = pix * CDIM + grp * 8;

    float4 a = *(const float4*)(x + base);
    float4 b = *(const float4*)(x + base + 4);
    float v[8] = {a.x, a.y, a.z, a.w, b.x, b.y, b.z, b.w};

    float mu = 0.f;
#pragma unroll
    for (int i = 0; i < 8; ++i) mu += v[i];
    mu *= 0.125f;
    float var = 0.f;
#pragma unroll
    for (int i = 0; i < 8; ++i) { float d = v[i] - mu; var += d * d; }
    var *= 0.125f;
    float inv = rsqrtf(var + EPSV);

    int cbase = grp * 8;
    unsigned short o[8];
#pragma unroll
    for (int i = 0; i < 8; ++i) {
        float y = (v[i] - mu) * inv;
        y = gamma[cbase + i] * y + beta[cbase + i];
        o[i] = f2bf(y);
    }
    uint4 packed = make_uint4((unsigned)o[0] | ((unsigned)o[1] << 16),
                              (unsigned)o[2] | ((unsigned)o[3] << 16),
                              (unsigned)o[4] | ((unsigned)o[5] << 16),
                              (unsigned)o[6] | ((unsigned)o[7] << 16));
    *(uint4*)(xn + base) = packed;
}

// ---------------------------------------------------------------------------
// WMMA GEMM: C[M,N] = scale * A[M,K] * Bt[N,K]^T  (+ resid)
// A, Bt bf16 bits row-major; fp32 accumulate.
// Block = 256 threads (8 waves); block tile 128(M) x 128(N);
// wave tile 64(M) x 32(N) = 4x2 WMMA tiles (8 WMMA / K-step).
// Bt slab (128x32 bf16 = 8KB) staged via global_load_async_to_lds_b128,
// double buffered (16KB dynamic LDS). A fragments software-pipelined; the
// K loop is unrolled 2x so fragment sets / LDS buffers ping-pong by name.
// grid = dim3(N/128, M/128). M%128==0, N%128==0, (K/32) even, K>=64.
// ---------------------------------------------------------------------------

// issue one K-step's WMMAs reading fragment set CUR; optionally stage the
// next B slab and load the next A fragment set. Branch-free.
#define LOADA(DST, KK)                                                     \
    DST[0] = load_frag_g(A, K, gm + ll,      (KK), lh);                    \
    DST[1] = load_frag_g(A, K, gm + 16 + ll, (KK), lh);                    \
    DST[2] = load_frag_g(A, K, gm + 32 + ll, (KK), lh);                    \
    DST[3] = load_frag_g(A, K, gm + 48 + ll, (KK), lh);

#define GEMM_STEP(CUR, BUFCUR, STAGE_STMT, LOAD_STMT)                      \
    {                                                                      \
        asm volatile("s_wait_asynccnt 0x0" ::: "memory");                  \
        __syncthreads();                                                   \
        STAGE_STMT;                                                        \
        const unsigned short* sb = bsh + (BUFCUR) * 4096;                  \
        v16bf b0 = load_frag_lds(sb, gnl + ll,      lh);                   \
        v16bf b1 = load_frag_lds(sb, gnl + 16 + ll, lh);                   \
        LOAD_STMT;                                                         \
        acc[0][0] = wmma_bf16(CUR[0], b0, acc[0][0]);                      \
        acc[0][1] = wmma_bf16(CUR[0], b1, acc[0][1]);                      \
        acc[1][0] = wmma_bf16(CUR[1], b0, acc[1][0]);                      \
        acc[1][1] = wmma_bf16(CUR[1], b1, acc[1][1]);                      \
        acc[2][0] = wmma_bf16(CUR[2], b0, acc[2][0]);                      \
        acc[2][1] = wmma_bf16(CUR[2], b1, acc[2][1]);                      \
        acc[3][0] = wmma_bf16(CUR[3], b0, acc[3][0]);                      \
        acc[3][1] = wmma_bf16(CUR[3], b1, acc[3][1]);                      \
    }

template <bool OUT_BF16, bool RESID>
__global__ __launch_bounds__(256)
void wmma_gemm_kernel(const unsigned short* __restrict__ A,
                      const unsigned short* __restrict__ Bt,
                      unsigned short* __restrict__ Cb,
                      float* __restrict__ Cf,
                      const float* __restrict__ resid,
                      int M, int N, int K, float scale)
{
    extern __shared__ unsigned short bsh[];          // 2 x (128x32) bf16, offset 0

    const int lane = threadIdx.x & 31;
    const int wid  = threadIdx.x >> 5;               // 0..7
    const int ll   = lane & 15;
    const int lh   = lane >> 4;

    const int gm  = blockIdx.y * 128 + (wid & 1) * 64;   // wave M origin
    const int gnl = (wid >> 1) * 32;                     // wave N origin in block
    const int gn  = blockIdx.x * 128 + gnl;
    const int bN  = blockIdx.x * 128;                    // block N origin

    // --- async stage of Bt[bN..bN+128, k0..k0+32) into LDS buffer `buf` ---
    // 512 x 16B chunks; each thread issues exactly two, no branches.
    const int srow = threadIdx.x >> 2;                   // 0..63
    const int scol = (threadIdx.x & 3) * 8;              // element col
    const unsigned int sbase = (unsigned int)(srow * 64 + scol * 2);
    auto stage = [&](int k0, int buf) {
        const unsigned short* g0 = Bt + (size_t)(bN + srow) * K + k0 + scol;
        const unsigned short* g1 = g0 + (size_t)64 * K;  // rows 64..127
        unsigned int l0 = (unsigned int)(buf * 8192) + sbase;
        asm volatile("global_load_async_to_lds_b128 %0, %1, off"
                     :: "v"(l0), "v"(g0) : "memory");
        asm volatile("global_load_async_to_lds_b128 %0, %1, off"
                     :: "v"(l0 + 4096u), "v"(g1) : "memory");
    };

    v8f acc[4][2] = {};
    v16bf af[4], an[4];

    stage(0, 0);
    LOADA(af, 0)

    // 2x-unrolled ping-pong main loop; handles K-steps up to K-96, leaving
    // exactly two steps (K-64, K-32) for the epilogue. (K/32) is even.
    int k0 = 0;
    for (; k0 + 64 < K; k0 += 64) {
        GEMM_STEP(af, 0, stage(k0 + 32, 1), LOADA(an, k0 + 32))
        GEMM_STEP(an, 1, stage(k0 + 64, 0), LOADA(af, k0 + 64))
    }
    // epilogue: steps K-64 (stage last slab, load last A set) and K-32.
    GEMM_STEP(af, 0, stage(K - 32, 1), LOADA(an, K - 32))
    GEMM_STEP(an, 1, (void)0, (void)0)

    // C/D layout: VGPR r, lanes 0-15 -> M=r, N=lane; lanes 16-31 -> M=8+r.
#pragma unroll
    for (int ti = 0; ti < 4; ++ti) {
#pragma unroll
        for (int tj = 0; tj < 2; ++tj) {
            int n = gn + tj * 16 + ll;
#pragma unroll
            for (int r = 0; r < 8; ++r) {
                int m = gm + ti * 16 + lh * 8 + r;
                float v = acc[ti][tj][r] * scale;
                size_t idx = (size_t)m * N + n;
                if (RESID) v += resid[idx];
                if (OUT_BF16) Cb[idx] = f2bf(v);
                else          Cf[idx] = v;
            }
        }
    }
}

// ---------------------------------------------------------------------------
// Kernel: per-batch repack of T (natural [h,w,c] layout) into the quirky
// reshaped operands:
//   Q[p,j]  = t[(256*(p&3)+j)>>5, (256*(p&3)+j)&31, p>>2]     (Q == V layout)
//   VT[j,p] = Q[p,j]
//   KT[m,i] = t[i>>3, 4*(i&7)+(m>>8), m&255]
// ---------------------------------------------------------------------------
__global__ void repack_kernel(const unsigned short* __restrict__ tb,  // [1024,256] slice
                              unsigned short* __restrict__ Q,         // [1024,256]
                              unsigned short* __restrict__ KT,        // [1024,256]
                              unsigned short* __restrict__ VT)        // [256,1024]
{
    int t = blockIdx.x * blockDim.x + threadIdx.x;
    int p = t >> 8;
    int j = t & 255;

    // q / v gather
    int r  = p & 3;
    int f  = 256 * r + j;
    int hh = f >> 5;
    int ww = f & 31;
    int cc = p >> 2;
    unsigned short qv = tb[((size_t)hh * 32 + ww) * 256 + cc];
    Q [(size_t)p * 256 + j]  = qv;
    VT[(size_t)j * 1024 + p] = qv;

    // k gather (KT[m,i] with m=p, i=j)
    int hh2 = j >> 3;
    int ww2 = ((j & 7) << 2) + (p >> 8);
    int cc2 = p & 255;
    KT[(size_t)p * 256 + j] = tb[((size_t)hh2 * 32 + ww2) * 256 + cc2];
}

// ---------------------------------------------------------------------------
// Kernel: row softmax of S[1024,1024] fp32, transposed bf16 store:
//   AT[m,p] = softmax_m(S[p,:])[m]
// One wave per row; 4 waves / block; 256 blocks.
// ---------------------------------------------------------------------------
__global__ void softmax_t_kernel(const float* __restrict__ S,
                                 unsigned short* __restrict__ AT)
{
    int lane = threadIdx.x & 31;
    int row  = blockIdx.x * 4 + (threadIdx.x >> 5);
    const float* srow = S + (size_t)row * HWDIM;

    float v[32];
    float mx = -3.402823466e+38f;
#pragma unroll
    for (int i = 0; i < 32; ++i) {
        v[i] = srow[lane + 32 * i];
        mx = fmaxf(mx, v[i]);
    }
#pragma unroll
    for (int s = 16; s >= 1; s >>= 1)
        mx = fmaxf(mx, __shfl_xor(mx, s, 32));

    float sum = 0.f;
#pragma unroll
    for (int i = 0; i < 32; ++i) {
        v[i] = __expf(v[i] - mx);
        sum += v[i];
    }
#pragma unroll
    for (int s = 16; s >= 1; s >>= 1)
        sum += __shfl_xor(sum, s, 32);
    float inv = 1.0f / sum;

#pragma unroll
    for (int i = 0; i < 32; ++i) {
        int m = lane + 32 * i;
        AT[(size_t)m * HWDIM + row] = f2bf(v[i] * inv);
    }
}

// ---------------------------------------------------------------------------
// Host-side launch
// ---------------------------------------------------------------------------
extern "C" void kernel_launch(void* const* d_in, const int* in_sizes, int n_in,
                              void* d_out, int out_size, void* d_ws, size_t ws_size,
                              hipStream_t stream) {
    const float* x     = (const float*)d_in[0];   // [32,32,32,256]
    const float* gamma = (const float*)d_in[1];   // [256]
    const float* beta  = (const float*)d_in[2];   // [256]
    const float* w_qkv = (const float*)d_in[3];   // [256,768]
    const float* w_out = (const float*)d_in[4];   // [256,256]
    float* out = (float*)d_out;                   // [32,32,32,256]

    // workspace layout (bytes)
    char* ws = (char*)d_ws;
    size_t off = 0;
    auto alloc = [&](size_t bytes) { char* p = ws + off; off += (bytes + 255) & ~(size_t)255; return p; };
    unsigned short* xn    = (unsigned short*)alloc((size_t)NPIX * CDIM * 2);   // 16 MB
    unsigned short* tbf   = (unsigned short*)alloc((size_t)NPIX * CDIM * 2);   // 16 MB
    unsigned short* wsumT = (unsigned short*)alloc((size_t)CDIM * CDIM * 2);
    unsigned short* woutT = (unsigned short*)alloc((size_t)CDIM * CDIM * 2);
    unsigned short* Qb    = (unsigned short*)alloc((size_t)HWDIM * CDIM * 2);  // per-batch
    unsigned short* KTb   = (unsigned short*)alloc((size_t)HWDIM * CDIM * 2);
    unsigned short* VTb   = (unsigned short*)alloc((size_t)CDIM * HWDIM * 2);
    float*          Sb    = (float*)alloc((size_t)HWDIM * HWDIM * 4);          // 4 MB, reused
    unsigned short* ATb   = (unsigned short*)alloc((size_t)HWDIM * HWDIM * 2); // 2 MB, reused
    unsigned short* Ovec  = (unsigned short*)alloc((size_t)NPIX * CDIM * 2);   // 16 MB
    (void)ws_size;

    const size_t lds_bytes = 2 * 128 * 32 * sizeof(unsigned short);  // 16 KB

    // 1) weights: fold (Wq+Wk+Wv)/3, transpose to [N,K] bf16
    prep_weights_kernel<<<256, 256, 0, stream>>>(w_qkv, w_out, wsumT, woutT);

    // 2) GroupNorm -> bf16
    groupnorm_kernel<<<(NPIX * 32) / 256, 256, 0, stream>>>(x, gamma, beta, xn);

    // 3) T = xn @ (Wq+Wk+Wv)/3   [32768,256] bf16 (natural [b,h,w,c] layout)
    wmma_gemm_kernel<true, false>
        <<<dim3(CDIM / 128, NPIX / 128), 256, lds_bytes, stream>>>(
        xn, wsumT, tbf, nullptr, nullptr, NPIX, CDIM, CDIM, 1.0f);

    // 4) per-batch attention
    for (int b = 0; b < BATCH; ++b) {
        const unsigned short* tslice = tbf + (size_t)b * HWDIM * CDIM;

        repack_kernel<<<(HWDIM * CDIM) / 256, 256, 0, stream>>>(tslice, Qb, KTb, VTb);

        // S = (1/16) * Q @ K   (Bt = K^T)
        wmma_gemm_kernel<false, false>
            <<<dim3(HWDIM / 128, HWDIM / 128), 256, lds_bytes, stream>>>(
            Qb, KTb, nullptr, Sb, nullptr, HWDIM, HWDIM, CDIM, SCALE_S);

        // AT = softmax(S, axis=-1)^T  (bf16)
        softmax_t_kernel<<<HWDIM / 4, 128, 0, stream>>>(Sb, ATb);

        // O = attn^T @ V   (A = AT [1024,1024], Bt = V^T [256,1024])
        wmma_gemm_kernel<true, false>
            <<<dim3(CDIM / 128, HWDIM / 128), 256, lds_bytes, stream>>>(
            ATb, VTb, Ovec + (size_t)b * HWDIM * CDIM, nullptr, nullptr,
            HWDIM, CDIM, HWDIM, 1.0f);
    }

    // 5) Y = O @ w_out + x  (fp32 out)
    wmma_gemm_kernel<false, true>
        <<<dim3(CDIM / 128, NPIX / 128), 256, lds_bytes, stream>>>(
        Ovec, woutT, nullptr, out, x, NPIX, CDIM, CDIM, 1.0f);
}
